// TransformerDecoder_46325517254776
// MI455X (gfx1250) — matrix operand order repaired and checked by reading, not compiled, not run on hardware
//
#include <hip/hip_runtime.h>
#include <hip/hip_bf16.h>
#include <math.h>

// ---------------------------------------------------------------------------
// Types for CDNA5 WMMA (wave32): V_WMMA_F32_16X16X32_BF16
// ---------------------------------------------------------------------------
typedef __bf16 bf16;
typedef __attribute__((ext_vector_type(16))) __bf16 v16bf;
typedef __attribute__((ext_vector_type(8)))  float  v8f;

union ABFrag { v16bf v; uint4 q[2]; };

__device__ __forceinline__ bf16 f2bf(float f) {
  union { float f; unsigned u; } a; a.f = f;
  unsigned r = a.u + 0x7FFFu + ((a.u >> 16) & 1u);   // round-to-nearest-even
  unsigned short h = (unsigned short)(r >> 16);
  return __builtin_bit_cast(bf16, h);
}

__device__ __forceinline__ v8f zero8() {
  v8f z;
  for (int i = 0; i < 8; i++) z[i] = 0.f;
  return z;
}

// ---------------------------------------------------------------------------
// CDNA5 async memory engine: global -> LDS DMA (no VGPR round trip), tracked
// by ASYNCcnt. Per-lane: LDS[vDst] = MEM[vAddr], 16 bytes.
// Flat pointers to __shared__ carry the LDS byte offset in addr[31:0]
// (flat LDS aperture, ISA 10.2), which is what VDST expects.
// ---------------------------------------------------------------------------
__device__ __forceinline__ void async_load_b128(const bf16* gsrc, bf16* ldst) {
  unsigned lds = (unsigned)(unsigned long long)ldst;
  asm volatile("global_load_async_to_lds_b128 %0, %1, off"
               :: "v"(lds), "v"(gsrc) : "memory");
}
__device__ __forceinline__ void wait_async0() {
  asm volatile("s_wait_asynccnt 0x0" ::: "memory");
}

// Problem constants
#define DM 1024
#define HH 16
#define DKK 64
#define FF 4096

// ---------------------------------------------------------------------------
// LayerNorm over D=1024 per row, fp32 in -> bf16 out. One 256-thread block/row.
// ---------------------------------------------------------------------------
__global__ __launch_bounds__(256)
void layernorm_kernel(const float* __restrict__ x, const float* __restrict__ g,
                      const float* __restrict__ bb, bf16* __restrict__ out)
{
  const int row = blockIdx.x;
  const float* xr = x + (size_t)row * DM;
  float4 v = ((const float4*)xr)[threadIdx.x];
  float s  = v.x + v.y + v.z + v.w;
  float s2 = v.x*v.x + v.y*v.y + v.z*v.z + v.w*v.w;
  for (int off = 16; off; off >>= 1) {
    s  += __shfl_xor(s,  off, 32);
    s2 += __shfl_xor(s2, off, 32);
  }
  __shared__ float rs[8], rs2[8];
  const int wid = threadIdx.x >> 5, lane = threadIdx.x & 31;
  if (lane == 0) { rs[wid] = s; rs2[wid] = s2; }
  __syncthreads();
  if (wid == 0) {
    float a  = (lane < 8) ? rs[lane]  : 0.f;
    float a2 = (lane < 8) ? rs2[lane] : 0.f;
    for (int off = 4; off; off >>= 1) {
      a  += __shfl_xor(a,  off, 32);
      a2 += __shfl_xor(a2, off, 32);
    }
    if (lane == 0) { rs[0] = a; rs2[0] = a2; }
  }
  __syncthreads();
  const float mean = rs[0] * (1.0f / DM);
  const float var  = rs2[0] * (1.0f / DM) - mean * mean;
  const float rstd = rsqrtf(var + 1e-5f);
  float4 gv = ((const float4*)g)[threadIdx.x];
  float4 bv = ((const float4*)bb)[threadIdx.x];
  union { bf16 h[4]; uint2 u; } pk;
  pk.h[0] = f2bf((v.x - mean) * rstd * gv.x + bv.x);
  pk.h[1] = f2bf((v.y - mean) * rstd * gv.y + bv.y);
  pk.h[2] = f2bf((v.z - mean) * rstd * gv.z + bv.z);
  pk.h[3] = f2bf((v.w - mean) * rstd * gv.w + bv.w);
  *(uint2*)(out + (size_t)row * DM + threadIdx.x * 4) = pk.u;
}

// ---------------------------------------------------------------------------
// Elementwise fp32 -> bf16
// ---------------------------------------------------------------------------
__global__ __launch_bounds__(256)
void convert_bf16_kernel(const float* __restrict__ in, bf16* __restrict__ out, int n)
{
  int i = blockIdx.x * 256 + threadIdx.x;
  if (i < n) out[i] = f2bf(in[i]);
}

// W[h][d][kk] (H,D,DK) -> out[(h*DK+kk)][d]  (i.e. [N][K] transposed, bf16)
__global__ __launch_bounds__(256)
void convert_wqkv_kernel(const float* __restrict__ in, bf16* __restrict__ out)
{
  int i = blockIdx.x * 256 + threadIdx.x;    // over N*K = (H*DK)*D
  if (i >= HH * DKK * DM) return;
  int d = i % DM;
  int n = i / DM;
  int h  = n >> 6;        // n / DK
  int kk = n & 63;        // n % DK
  out[i] = f2bf(in[((size_t)h * DM + d) * DKK + kk]);
}

// W[k][n] (K,N) -> out[n][k]  ([N][K] transposed, bf16)
__global__ __launch_bounds__(256)
void convert_wT_kernel(const float* __restrict__ in, bf16* __restrict__ out, int K, int N)
{
  int i = blockIdx.x * 256 + threadIdx.x;    // over N*K
  if (i >= K * N) return;
  int k = i % K;
  int n = i / K;
  out[i] = f2bf(in[(size_t)k * N + n]);
}

// ---------------------------------------------------------------------------
// bf16 GEMM: C[M,N] = A[M,K] * Bt[N,K]^T + bias[N]  (WMMA f32_16x16x32_bf16)
// Block tile 128x128x32. Tiles staged with GLOBAL_LOAD_ASYNC_TO_LDS_B128 and
// double-buffered: the DMA for K-step k+1 overlaps the 8 WMMAs of step k.
// One barrier + one s_wait_asynccnt per K-step.
// EPI 0: bf16 out.  EPI 1: fp32 out = resid + (acc+bias).  EPI 2: gelu, bf16 out.
// ---------------------------------------------------------------------------
template<int EPI>
__global__ __launch_bounds__(256)
void gemm_bf16_kernel(const bf16* __restrict__ A, const bf16* __restrict__ Bt,
                      const float* __restrict__ bias, const float* __restrict__ resid,
                      bf16* __restrict__ outb, float* __restrict__ outf,
                      int M, int N, int K)
{
  __shared__ __align__(16) bf16 lA[2][128 * 32];
  __shared__ __align__(16) bf16 lB[2][128 * 32];

  const int tid  = threadIdx.x;
  const int lane = tid & 31;
  const int wid  = tid >> 5;
  const int lg   = lane >> 4;    // half-wave group
  const int ln   = lane & 15;
  const int wm   = wid & 3;      // wave row   (4 x 32 rows)
  const int wn   = wid >> 2;     // wave col   (2 x 64 cols)
  const int bm   = blockIdx.x * 128;
  const int bn   = blockIdx.y * 128;

  const int lrow = tid >> 1;           // 0..127  (tile staging)
  const int lcol = (tid & 1) * 16;     // 0 or 16

  v8f acc[2][4];
  for (int i = 0; i < 2; i++)
    for (int j = 0; j < 4; j++) acc[i][j] = zero8();

  const bf16* gA = A  + (size_t)(bm + lrow) * K + lcol;
  const bf16* gB = Bt + (size_t)(bn + lrow) * K + lcol;

  auto stage = [&](int buf, int k0) {
    async_load_b128(gA + k0,     &lA[buf][lrow * 32 + lcol]);
    async_load_b128(gA + k0 + 8, &lA[buf][lrow * 32 + lcol + 8]);
    async_load_b128(gB + k0,     &lB[buf][lrow * 32 + lcol]);
    async_load_b128(gB + k0 + 8, &lB[buf][lrow * 32 + lcol + 8]);
  };

  stage(0, 0);
  int cur = 0;
  for (int k0 = 0; k0 < K; k0 += 32) {
    wait_async0();      // buffer `cur` fully written by the async engine
    __syncthreads();    // ...and every wave has finished reading `cur^1`
    if (k0 + 32 < K) stage(cur ^ 1, k0 + 32);   // DMA overlaps WMMAs below

    ABFrag fa[2], fb[4];
#pragma unroll
    for (int mt = 0; mt < 2; mt++) {
      int ar = wm * 32 + mt * 16 + ln;
      fa[mt].q[0] = *(const uint4*)&lA[cur][ar * 32 + lg * 8];
      fa[mt].q[1] = *(const uint4*)&lA[cur][ar * 32 + lg * 8 + 16];
    }
#pragma unroll
    for (int nt = 0; nt < 4; nt++) {
      int br = wn * 64 + nt * 16 + ln;
      fb[nt].q[0] = *(const uint4*)&lB[cur][br * 32 + lg * 8];
      fb[nt].q[1] = *(const uint4*)&lB[cur][br * 32 + lg * 8 + 16];
    }
#pragma unroll
    for (int mt = 0; mt < 2; mt++)
#pragma unroll
      for (int nt = 0; nt < 4; nt++)
        acc[mt][nt] = __builtin_amdgcn_wmma_f32_16x16x32_bf16(
            false, fa[mt].v, false, fb[nt].v, (short)0, acc[mt][nt], false, false);

    cur ^= 1;
  }

  for (int mt = 0; mt < 2; mt++) {
    for (int nt = 0; nt < 4; nt++) {
      const int col = bn + wn * 64 + nt * 16 + ln;
      const float bv = bias[col];
      for (int r = 0; r < 8; r++) {
        const int row = bm + wm * 32 + mt * 16 + lg * 8 + r;
        float v = acc[mt][nt][r] + bv;
        const size_t idx = (size_t)row * N + col;
        if (EPI == 0) {
          outb[idx] = f2bf(v);
        } else if (EPI == 1) {
          outf[idx] = resid[idx] + v;
        } else {
          float gel = 0.5f * v * (1.0f + erff(v * 0.70710678118f));
          outb[idx] = f2bf(gel);
        }
      }
    }
  }
}

// ---------------------------------------------------------------------------
// Flash attention: one wave per (b, h, 16 query rows). Q/K/V bf16 [B,*,H,DK].
// Processes keys 32 at a time: QK^T (4 WMMA), online softmax, P*V (4 WMMA).
// causal!=0 -> lower-triangular mask; kmask[b*Sk+key]==0 -> masked key.
// ---------------------------------------------------------------------------
__global__ __launch_bounds__(32)
void flash_attn_kernel(const bf16* __restrict__ Q, const bf16* __restrict__ Kv,
                       const bf16* __restrict__ V, const int* __restrict__ kmask,
                       bf16* __restrict__ O, int Tq, int Sk, int causal)
{
  __shared__ __align__(16) bf16 lP[16 * 32];
  __shared__ __align__(16) bf16 lV[64 * 32];

  const int lane = threadIdx.x;
  const int lg = lane >> 4;
  const int ln = lane & 15;
  const int t0 = blockIdx.x * 16;
  const int h  = blockIdx.y;
  const int b  = blockIdx.z;
  const float scale = 0.125f;  // 1/sqrt(64)

  // Q fragments (A operand), dk 0..31 and 32..63
  const bf16* qp = Q + (((size_t)b * Tq + (t0 + ln)) * HH + h) * DKK;
  const int ach = lg * 8;
  ABFrag fq0, fq1;
  fq0.q[0] = *(const uint4*)(qp + ach);
  fq0.q[1] = *(const uint4*)(qp + ach + 16);
  fq1.q[0] = *(const uint4*)(qp + 32 + ach);
  fq1.q[1] = *(const uint4*)(qp + 32 + ach + 16);

  float m[8], l[8];
  v8f acc[4];
  for (int r = 0; r < 8; r++) { m[r] = -3.0e38f; l[r] = 0.f; }
  for (int nt = 0; nt < 4; nt++) acc[nt] = zero8();

  const int jmax = causal ? (t0 + 16) : Sk;   // keys needed: key <= t0+15
  for (int jb = 0; jb < jmax; jb += 32) {
    // ---- scores: S = Q * K^T for 32 keys (two 16-wide tiles) ----
    v8f s[2];
    int masked[2];
#pragma unroll
    for (int nt = 0; nt < 2; nt++) {
      s[nt] = zero8();
      const int key = jb + nt * 16 + ln;
      masked[nt] = (kmask[(size_t)b * Sk + key] == 0);
      const bf16* kp = Kv + (((size_t)b * Sk + key) * HH + h) * DKK;
      const int kb = lg * 8;
      ABFrag fk0, fk1;
      fk0.q[0] = *(const uint4*)(kp + kb);
      fk0.q[1] = *(const uint4*)(kp + kb + 16);
      fk1.q[0] = *(const uint4*)(kp + 32 + kb);
      fk1.q[1] = *(const uint4*)(kp + 32 + kb + 16);
      s[nt] = __builtin_amdgcn_wmma_f32_16x16x32_bf16(
          false, fq0.v, false, fk0.v, (short)0, s[nt], false, false);
      s[nt] = __builtin_amdgcn_wmma_f32_16x16x32_bf16(
          false, fq1.v, false, fk1.v, (short)0, s[nt], false, false);
    }

    // ---- online softmax (row = half-wave group), write P to LDS ----
    const int key0 = jb + ln, key1 = jb + 16 + ln;
    for (int r = 0; r < 8; r++) {
      const int trow = t0 + lg * 8 + r;
      const int msk0 = masked[0] || (causal && key0 > trow);
      const int msk1 = masked[1] || (causal && key1 > trow);
      float v0 = msk0 ? -1e30f : s[0][r] * scale;
      float v1 = msk1 ? -1e30f : s[1][r] * scale;
      float mx = fmaxf(v0, v1);
      for (int off = 8; off; off >>= 1) mx = fmaxf(mx, __shfl_xor(mx, off, 32));
      const float mnew  = fmaxf(m[r], mx);
      const float alpha = __expf(m[r] - mnew);
      const float p0 = msk0 ? 0.f : __expf(v0 - mnew);
      const float p1 = msk1 ? 0.f : __expf(v1 - mnew);
      float rsum = p0 + p1;
      for (int off = 8; off; off >>= 1) rsum += __shfl_xor(rsum, off, 32);
      l[r] = l[r] * alpha + rsum;
      m[r] = mnew;
      for (int nt = 0; nt < 4; nt++) acc[nt][r] *= alpha;
      const int rowl = lg * 8 + r;
      lP[rowl * 32 + ln]      = f2bf(p0);
      lP[rowl * 32 + 16 + ln] = f2bf(p1);
    }

    // ---- stage V tile (32 keys x 64 dk) transposed -> lV[dk][key] ----
    const bf16* vp = V + (((size_t)b * Sk + jb) * HH + h) * DKK;
    for (int i = lane; i < 256; i += 32) {
      const int keyl = i >> 3;
      const int dkb  = (i & 7) * 8;
      uint4 vv = *(const uint4*)(vp + (size_t)keyl * (HH * DKK) + dkb);
      const bf16* ve = (const bf16*)&vv;
      for (int e = 0; e < 8; e++) lV[(dkb + e) * 32 + keyl] = ve[e];
    }
    __syncthreads();

    // ---- acc += P * V  (4 WMMA over dk tiles) ----
    ABFrag fp;
    fp.q[0] = *(const uint4*)&lP[ln * 32 + lg * 8];
    fp.q[1] = *(const uint4*)&lP[ln * 32 + lg * 8 + 16];
#pragma unroll
    for (int nt = 0; nt < 4; nt++) {
      const int dk = nt * 16 + ln;
      ABFrag fv;
      fv.q[0] = *(const uint4*)&lV[dk * 32 + lg * 8];
      fv.q[1] = *(const uint4*)&lV[dk * 32 + lg * 8 + 16];
      acc[nt] = __builtin_amdgcn_wmma_f32_16x16x32_bf16(
          false, fp.v, false, fv.v, (short)0, acc[nt], false, false);
    }
    __syncthreads();
  }

  // ---- normalize and store O[b, t, h, dk] bf16 ----
  for (int r = 0; r < 8; r++) {
    const int trow = t0 + lg * 8 + r;
    const float inv = (l[r] > 0.f) ? (1.0f / l[r]) : 0.f;
    bf16* op = O + (((size_t)b * Tq + trow) * HH + h) * DKK;
    for (int nt = 0; nt < 4; nt++)
      op[nt * 16 + ln] = f2bf(acc[nt][r] * inv);
  }
}

// ---------------------------------------------------------------------------
// Host-side orchestration
// ---------------------------------------------------------------------------
extern "C" void kernel_launch(void* const* d_in, const int* in_sizes, int n_in,
                              void* d_out, int out_size, void* d_ws, size_t ws_size,
                              hipStream_t stream)
{
  const float* x     = (const float*)d_in[0];
  const float* enc   = (const float*)d_in[1];
  const int*   smask = (const int*)d_in[2];
  const int*   cmask = (const int*)d_in[3];
  const float* ln1g  = (const float*)d_in[4];
  const float* ln1b  = (const float*)d_in[5];
  const float* ln2g  = (const float*)d_in[6];
  const float* ln2b  = (const float*)d_in[7];
  const float* ln3g  = (const float*)d_in[8];
  const float* ln3b  = (const float*)d_in[9];
  const float* Wq_s  = (const float*)d_in[10];
  const float* bq_s  = (const float*)d_in[11];
  const float* Wk_s  = (const float*)d_in[12];
  const float* bk_s  = (const float*)d_in[13];
  const float* Wv_s  = (const float*)d_in[14];
  const float* bv_s  = (const float*)d_in[15];
  const float* Wo_s  = (const float*)d_in[16];
  const float* bo_s  = (const float*)d_in[17];
  const float* Wq_c  = (const float*)d_in[18];
  const float* bq_c  = (const float*)d_in[19];
  const float* Wk_c  = (const float*)d_in[20];
  const float* bk_c  = (const float*)d_in[21];
  const float* Wv_c  = (const float*)d_in[22];
  const float* bv_c  = (const float*)d_in[23];
  const float* Wo_c  = (const float*)d_in[24];
  const float* bo_c  = (const float*)d_in[25];
  const float* W1    = (const float*)d_in[26];
  const float* b1    = (const float*)d_in[27];
  const float* W2    = (const float*)d_in[28];
  const float* b2    = (const float*)d_in[29];

  const int Bb = 4, Tt = 1024, Ss = 1024;
  const int BT = Bb * Tt, BS = Bb * Ss;

  char* wsp = (char*)d_ws;
  auto carve = [&](size_t bytes) -> void* {
    void* p = (void*)wsp;
    wsp += (bytes + 255) & ~(size_t)255;
    return p;
  };

  bf16*  xn   = (bf16*) carve((size_t)BT * DM * 2);
  bf16*  encb = (bf16*) carve((size_t)BS * DM * 2);
  bf16*  qb   = (bf16*) carve((size_t)BT * DM * 2);
  bf16*  kb   = (bf16*) carve((size_t)BS * DM * 2);
  bf16*  vb   = (bf16*) carve((size_t)BS * DM * 2);
  bf16*  ab   = (bf16*) carve((size_t)BT * DM * 2);
  float* x1   = (float*)carve((size_t)BT * DM * 4);
  float* x2   = (float*)carve((size_t)BT * DM * 4);
  bf16*  wq   = (bf16*) carve((size_t)DM * DM * 2);
  bf16*  wk   = (bf16*) carve((size_t)DM * DM * 2);
  bf16*  wv   = (bf16*) carve((size_t)DM * DM * 2);
  bf16*  wo   = (bf16*) carve((size_t)DM * DM * 2);
  bf16*  w1t  = (bf16*) carve((size_t)DM * FF * 2);
  bf16*  w2t  = (bf16*) carve((size_t)FF * DM * 2);
  bf16*  hb   = (bf16*) carve((size_t)BT * FF * 2);

  const dim3 gProj(BT / 128, DM / 128);   // M=4096, N=1024
  const dim3 gF1(BT / 128, FF / 128);     // M=4096, N=4096
  const dim3 gAttn(Tt / 16, HH, Bb);
  const int cvt1M = (DM * DM + 255) / 256;

  // ===== Block 1: self-attention =====
  layernorm_kernel<<<BT, 256, 0, stream>>>(x, ln1g, ln1b, xn);
  convert_wqkv_kernel<<<cvt1M, 256, 0, stream>>>(Wq_s, wq);
  convert_wqkv_kernel<<<cvt1M, 256, 0, stream>>>(Wk_s, wk);
  convert_wqkv_kernel<<<cvt1M, 256, 0, stream>>>(Wv_s, wv);
  convert_wT_kernel<<<cvt1M, 256, 0, stream>>>(Wo_s, wo, DM, DM);
  gemm_bf16_kernel<0><<<gProj, 256, 0, stream>>>(xn, wq, bq_s, nullptr, qb, nullptr, BT, DM, DM);
  gemm_bf16_kernel<0><<<gProj, 256, 0, stream>>>(xn, wk, bk_s, nullptr, kb, nullptr, BT, DM, DM);
  gemm_bf16_kernel<0><<<gProj, 256, 0, stream>>>(xn, wv, bv_s, nullptr, vb, nullptr, BT, DM, DM);
  flash_attn_kernel<<<gAttn, 32, 0, stream>>>(qb, kb, vb, smask, ab, Tt, Tt, 1);
  gemm_bf16_kernel<1><<<gProj, 256, 0, stream>>>(ab, wo, bo_s, x, nullptr, x1, BT, DM, DM);

  // ===== Block 2: cross-attention =====
  layernorm_kernel<<<BT, 256, 0, stream>>>(x1, ln2g, ln2b, xn);
  convert_bf16_kernel<<<(BS * DM + 255) / 256, 256, 0, stream>>>(enc, encb, BS * DM);
  convert_wqkv_kernel<<<cvt1M, 256, 0, stream>>>(Wq_c, wq);
  convert_wqkv_kernel<<<cvt1M, 256, 0, stream>>>(Wk_c, wk);
  convert_wqkv_kernel<<<cvt1M, 256, 0, stream>>>(Wv_c, wv);
  convert_wT_kernel<<<cvt1M, 256, 0, stream>>>(Wo_c, wo, DM, DM);
  gemm_bf16_kernel<0><<<gProj, 256, 0, stream>>>(xn,   wq, bq_c, nullptr, qb, nullptr, BT, DM, DM);
  gemm_bf16_kernel<0><<<dim3(BS / 128, DM / 128), 256, 0, stream>>>(encb, wk, bk_c, nullptr, kb, nullptr, BS, DM, DM);
  gemm_bf16_kernel<0><<<dim3(BS / 128, DM / 128), 256, 0, stream>>>(encb, wv, bv_c, nullptr, vb, nullptr, BS, DM, DM);
  flash_attn_kernel<<<gAttn, 32, 0, stream>>>(qb, kb, vb, cmask, ab, Tt, Ss, 0);
  gemm_bf16_kernel<1><<<gProj, 256, 0, stream>>>(ab, wo, bo_c, x1, nullptr, x2, BT, DM, DM);

  // ===== Block 3: FFN =====
  layernorm_kernel<<<BT, 256, 0, stream>>>(x2, ln3g, ln3b, xn);
  convert_wT_kernel<<<(DM * FF + 255) / 256, 256, 0, stream>>>(W1, w1t, DM, FF);
  convert_wT_kernel<<<(FF * DM + 255) / 256, 256, 0, stream>>>(W2, w2t, FF, DM);
  gemm_bf16_kernel<2><<<gF1, 256, 0, stream>>>(xn, w1t, b1, nullptr, hb, nullptr, BT, FF, DM);
  gemm_bf16_kernel<1><<<gProj, 256, 0, stream>>>(hb, w2t, b2, x2, nullptr, (float*)d_out, BT, DM, FF);
}
